// DCLMBlock_16329465660000
// MI455X (gfx1250) — matrix operand order repaired and checked by reading, not compiled, not run on hardware
//
#include <hip/hip_runtime.h>
#include <cstdint>
#include <cstddef>

// ---- problem constants ----
#define B_     4
#define S_     2048
#define H_     1536
#define NH_    12
#define HD_    128
#define K4     4
#define INNER_ 6144
#define MTOT   (B_*S_)          // 8192 token-rows
#define EPSF   1e-6f

typedef __bf16 bf16;
typedef float v8f  __attribute__((ext_vector_type(8)));
typedef bf16  v16bf __attribute__((ext_vector_type(16)));
typedef int   i4   __attribute__((ext_vector_type(4)));

struct alignas(16) U4 { unsigned int a, b, c, d; };
union Frag { v16bf v; U4 q[2]; };

// ---- gfx1250 async load-to-LDS (ASYNCcnt) availability probe ----
#if defined(__has_builtin)
#  if __has_builtin(__builtin_amdgcn_global_load_async_to_lds_b128) && \
      __has_builtin(__builtin_amdgcn_s_wait_asynccnt)
#    define USE_ASYNC_LDS 1
#  endif
#endif
#ifndef USE_ASYNC_LDS
#  define USE_ASYNC_LDS 0
#endif

#define ASG __attribute__((address_space(1)))
#define ASL __attribute__((address_space(3)))
static __device__ __forceinline__ ASG i4* gp4(const void* p) {
    // flat global address == AS1 address
    return (ASG i4*)(unsigned long long)(uintptr_t)p;
}
static __device__ __forceinline__ ASL i4* lp4(void* p) {
    // generic LDS pointer: low 32 bits are the LDS byte offset (ISA 10.2)
    return (ASL i4*)(unsigned int)(uintptr_t)p;
}

__constant__ int c_conv_dils[6] = {1, 2, 4, 8, 16, 32};
__constant__ int c_head_dils[12][3] = {
    {1,2,4},{1,1,1},{4,8,16},{8,16,32},{32,64,128},{64,128,256},
    {256,512,1024},{1,100,200},{1,500,1000},{1,1024,2048},{3,9,27},{5,25,125}};

__device__ __forceinline__ float sigf(float v) { return 1.f / (1.f + expf(-v)); }
__device__ __forceinline__ float gelu_exact(float v) {
    return 0.5f * v * (1.f + erff(v * 0.70710678118654752f));
}

// ============================================================================
// 1) Fused RMSNorm x3 + path gates + router: one block per token.
//    Also initializes out = x (paths accumulate into out afterwards).
// ============================================================================
__global__ void norm_gates_kernel(const float* __restrict__ x,
                                  const float* __restrict__ norm_w,
                                  const float* __restrict__ pg_w,
                                  const float* __restrict__ pg_b,
                                  const float* __restrict__ router_w,
                                  const float* __restrict__ router_b,
                                  float* __restrict__ out,
                                  float* __restrict__ n1f,
                                  bf16* __restrict__ n2b,
                                  bf16* __restrict__ n3b,
                                  float* __restrict__ gates,   // [3][MTOT]
                                  float* __restrict__ hw)      // [MTOT][NH]
{
    const int tok = blockIdx.x;
    const int tid = threadIdx.x;
    const size_t base = (size_t)tok * H_;
    __shared__ float red[256];
    __shared__ float red2[15][256];

    float ss = 0.f;
    for (int h = tid; h < H_; h += 256) { float v = x[base + h]; ss += v * v; }
    red[tid] = ss;
    __syncthreads();
    for (int off = 128; off > 0; off >>= 1) {
        if (tid < off) red[tid] += red[tid + off];
        __syncthreads();
    }
    const float inv = rsqrtf(red[0] * (1.f / (float)H_) + EPSF);

    float p[15];
#pragma unroll
    for (int v = 0; v < 15; ++v) p[v] = 0.f;

    for (int h = tid; h < H_; h += 256) {
        const float xv = x[base + h];
        const float xn = xv * inv;
        const float n1 = xn * norm_w[h];
        const float n2 = xn * norm_w[H_ + h];
        const float n3 = xn * norm_w[2 * H_ + h];
        out[base + h] = xv;
        n1f[base + h] = n1;
        n2b[base + h] = (bf16)n2;
        n3b[base + h] = (bf16)n3;
        p[0] += n1 * pg_w[h];
        p[1] += n2 * pg_w[H_ + h];
        p[2] += n3 * pg_w[2 * H_ + h];
#pragma unroll
        for (int r = 0; r < 12; ++r) p[3 + r] += n2 * router_w[r * H_ + h];
    }
#pragma unroll
    for (int v = 0; v < 15; ++v) red2[v][tid] = p[v];
    __syncthreads();
    if (tid < 15) {
        float s = 0.f;
        for (int j = 0; j < 256; ++j) s += red2[tid][j];
        if (tid < 3) gates[tid * MTOT + tok] = sigf(s + pg_b[tid]);
        else         hw[(size_t)tok * NH_ + (tid - 3)] = sigf(s + router_b[tid - 3]);
    }
}

// ============================================================================
// 2) f32 -> bf16 conversion (weights)
// ============================================================================
__global__ void f32_to_bf16_kernel(const float* __restrict__ src,
                                   bf16* __restrict__ dst, size_t n)
{
    size_t i = (size_t)blockIdx.x * 256 + threadIdx.x;
    const size_t stride = (size_t)gridDim.x * 256;
    for (; i < n; i += stride) dst[i] = (bf16)src[i];
}

// ============================================================================
// 3) Dilated conv stack (6 layers, GELU residual) — one block per (b, channel),
//    whole sequence lives in LDS ping-pong buffers (16 KB).
// ============================================================================
__global__ void conv_stack_kernel(const float* __restrict__ n1f,
                                  const float* __restrict__ cw,   // [6,H,1,K]
                                  const float* __restrict__ cb,   // [6,H]
                                  bf16* __restrict__ convA)       // [MTOT,H] bf16
{
    const int blk = blockIdx.x;
    const int b = blk / H_;
    const int c = blk % H_;
    const int tid = threadIdx.x;
    __shared__ float hb[2][S_];
    const size_t rowbase = (size_t)b * S_;

    for (int s = tid; s < S_; s += 256)
        hb[0][s] = n1f[(rowbase + s) * H_ + c];

    int cur = 0;
    for (int L = 0; L < 6; ++L) {
        const int d = c_conv_dils[L];
        const size_t wb = ((size_t)L * H_ + c) * K4;
        const float w0 = cw[wb + 0], w1 = cw[wb + 1], w2 = cw[wb + 2], w3 = cw[wb + 3];
        const float bias = cb[(size_t)L * H_ + c];
        __syncthreads();
        for (int s = tid; s < S_; s += 256) {
            float a = bias;
            const int i0 = s - 3 * d, i1 = s - 2 * d, i2 = s - d;
            a += w0 * (i0 >= 0 ? hb[cur][i0] : 0.f);
            a += w1 * (i1 >= 0 ? hb[cur][i1] : 0.f);
            a += w2 * (i2 >= 0 ? hb[cur][i2] : 0.f);
            a += w3 * hb[cur][s];
            hb[1 - cur][s] = hb[cur][s] + gelu_exact(a);
        }
        cur ^= 1;
    }
    __syncthreads();
    for (int s = tid; s < S_; s += 256)
        convA[(rowbase + s) * H_ + c] = (bf16)hb[cur][s];
}

// ============================================================================
// 4) Per-head dilated state convs. Fuses the GLU of the gate GEMM output on
//    load, 3 residual conv layers, then scales by router weight. One block
//    per (b, channel).
// ============================================================================
__global__ void head_conv_kernel(const bf16* __restrict__ g,     // [MTOT, 2H]
                                 const float* __restrict__ hwts, // [NH,3,HD,1,K]
                                 const float* __restrict__ hbs,  // [NH,3,HD]
                                 const float* __restrict__ hw,   // [MTOT,NH]
                                 bf16* __restrict__ state)       // [MTOT,H]
{
    const int blk = blockIdx.x;
    const int b = blk / H_;
    const int c = blk % H_;
    const int head = c >> 7;
    const int hd = c & (HD_ - 1);
    const int tid = threadIdx.x;
    __shared__ float hb[2][S_];
    const size_t rowbase = (size_t)b * S_;

    for (int s = tid; s < S_; s += 256) {
        const float cont = (float)g[(rowbase + s) * (2 * H_) + c];
        const float gt   = (float)g[(rowbase + s) * (2 * H_) + H_ + c];
        hb[0][s] = cont * sigf(gt);
    }
    int cur = 0;
    for (int j = 0; j < 3; ++j) {
        const int d = c_head_dils[head][j];
        const size_t wb = (((size_t)head * 3 + j) * HD_ + hd) * K4;
        const float w0 = hwts[wb + 0], w1 = hwts[wb + 1], w2 = hwts[wb + 2], w3 = hwts[wb + 3];
        const float bias = hbs[((size_t)head * 3 + j) * HD_ + hd];
        __syncthreads();
        for (int s = tid; s < S_; s += 256) {
            float a = bias;
            const int i0 = s - 3 * d, i1 = s - 2 * d, i2 = s - d;
            a += w0 * (i0 >= 0 ? hb[cur][i0] : 0.f);
            a += w1 * (i1 >= 0 ? hb[cur][i1] : 0.f);
            a += w2 * (i2 >= 0 ? hb[cur][i2] : 0.f);
            a += w3 * hb[cur][s];
            hb[1 - cur][s] = hb[cur][s] + a;
        }
        cur ^= 1;
    }
    __syncthreads();
    for (int s = tid; s < S_; s += 256)
        state[(rowbase + s) * H_ + c] =
            (bf16)(hb[cur][s] * hw[(rowbase + s) * NH_ + head]);
}

// ============================================================================
// 5) WMMA bf16 GEMM: out[M,N] = A[M,K] @ B[N,K]^T, fused epilogues.
//    Block = 256 threads (8 waves), 128x128 C tile, BK=32 K-steps.
//    Double-buffered LDS tiles staged with GLOBAL_LOAD_ASYNC_TO_LDS_B128
//    (ASYNCcnt) when available, else register-staged copies.
//    DUAL variant accumulates A against two weight halves (fused FFN GLU).
// ============================================================================
#define BM  128
#define BN  128
#define BKT 32
#define LDT 40   // LDS row pitch in bf16 elems (BK + 8 pad, 80B keeps 16B align)

enum { EPI_STORE_BF16 = 0, EPI_BIAS_GATE_ADD = 1, EPI_MULSIG_BF16 = 2, EPI_GLU = 3 };

template <int EPI, bool DUAL, bool HAS_BIAS>
__global__ void __launch_bounds__(256)
gemm_bf16_k(const bf16* __restrict__ A, const bf16* __restrict__ Bw,
            const bf16* __restrict__ Bw2,
            int M, int N, int Kd,
            float* __restrict__ Co, const float* __restrict__ gate,
            const float* __restrict__ bias,
            bf16* __restrict__ Cb, const bf16* __restrict__ Aux)
{
    __shared__ alignas(16) bf16 As[2][BM * LDT];
    __shared__ alignas(16) bf16 Bs[2][BN * LDT];
    __shared__ alignas(16) bf16 Bs2[DUAL ? 2 : 1][BN * LDT];

    const int tid  = threadIdx.x;
    const int lane = tid & 31;
    const int wave = tid >> 5;
    const int wm   = wave >> 2;      // 0..1 -> M offset wm*64
    const int wn   = wave & 3;       // 0..3 -> N offset wn*32
    const int hlf  = lane >> 4;      // lane half (ISA 7.12.2 layouts)
    const int ln   = lane & 15;
    const int blockM = blockIdx.y * BM;
    const int blockN = blockIdx.x * BN;

    // per-thread staging chunks: rows (tid>>2) and (tid>>2)+64, col (tid&3)*8
    const int row0 = tid >> 2;
    const int kc0  = (tid & 3) * 8;

    v8f acc[4][2], acc2[DUAL ? 4 : 1][2];
#pragma unroll
    for (int i = 0; i < 4; ++i)
#pragma unroll
        for (int j = 0; j < 2; ++j)
#pragma unroll
            for (int e = 0; e < 8; ++e) {
                acc[i][j][e] = 0.f;
                if constexpr (DUAL) acc2[i][j][e] = 0.f;
            }

#if USE_ASYNC_LDS
    // -------- async prologue: stage tile k0=0 into buffer 0 --------
    {
        __builtin_amdgcn_global_load_async_to_lds_b128(
            gp4(A + (size_t)(blockM + row0) * Kd + kc0),
            lp4(&As[0][row0 * LDT + kc0]), 0, 0);
        __builtin_amdgcn_global_load_async_to_lds_b128(
            gp4(A + (size_t)(blockM + row0 + 64) * Kd + kc0),
            lp4(&As[0][(row0 + 64) * LDT + kc0]), 0, 0);
        __builtin_amdgcn_global_load_async_to_lds_b128(
            gp4(Bw + (size_t)(blockN + row0) * Kd + kc0),
            lp4(&Bs[0][row0 * LDT + kc0]), 0, 0);
        __builtin_amdgcn_global_load_async_to_lds_b128(
            gp4(Bw + (size_t)(blockN + row0 + 64) * Kd + kc0),
            lp4(&Bs[0][(row0 + 64) * LDT + kc0]), 0, 0);
        if constexpr (DUAL) {
            __builtin_amdgcn_global_load_async_to_lds_b128(
                gp4(Bw2 + (size_t)(blockN + row0) * Kd + kc0),
                lp4(&Bs2[0][row0 * LDT + kc0]), 0, 0);
            __builtin_amdgcn_global_load_async_to_lds_b128(
                gp4(Bw2 + (size_t)(blockN + row0 + 64) * Kd + kc0),
                lp4(&Bs2[0][(row0 + 64) * LDT + kc0]), 0, 0);
        }
        __builtin_amdgcn_s_wait_asynccnt(0);
        __syncthreads();
    }
#else
    // -------- fallback prologue: register-staged copy of tile k0=0 --------
    {
        U4 ra0 = *reinterpret_cast<const U4*>(A  + (size_t)(blockM + row0) * Kd + kc0);
        U4 ra1 = *reinterpret_cast<const U4*>(A  + (size_t)(blockM + row0 + 64) * Kd + kc0);
        U4 rb0 = *reinterpret_cast<const U4*>(Bw + (size_t)(blockN + row0) * Kd + kc0);
        U4 rb1 = *reinterpret_cast<const U4*>(Bw + (size_t)(blockN + row0 + 64) * Kd + kc0);
        *reinterpret_cast<U4*>(&As[0][row0 * LDT + kc0]) = ra0;
        *reinterpret_cast<U4*>(&As[0][(row0 + 64) * LDT + kc0]) = ra1;
        *reinterpret_cast<U4*>(&Bs[0][row0 * LDT + kc0]) = rb0;
        *reinterpret_cast<U4*>(&Bs[0][(row0 + 64) * LDT + kc0]) = rb1;
        if constexpr (DUAL) {
            U4 rc0 = *reinterpret_cast<const U4*>(Bw2 + (size_t)(blockN + row0) * Kd + kc0);
            U4 rc1 = *reinterpret_cast<const U4*>(Bw2 + (size_t)(blockN + row0 + 64) * Kd + kc0);
            *reinterpret_cast<U4*>(&Bs2[0][row0 * LDT + kc0]) = rc0;
            *reinterpret_cast<U4*>(&Bs2[0][(row0 + 64) * LDT + kc0]) = rc1;
        }
        __syncthreads();
    }
#endif

    int cur = 0;
    for (int k0 = 0; k0 < Kd; k0 += BKT) {
        const int nxt = cur ^ 1;
        const bool more = (k0 + BKT < Kd);
        const int kn = k0 + BKT;

#if USE_ASYNC_LDS
        if (more) {  // issue next tile's async loads; they overlap the WMMAs
            __builtin_amdgcn_global_load_async_to_lds_b128(
                gp4(A + (size_t)(blockM + row0) * Kd + kn + kc0),
                lp4(&As[nxt][row0 * LDT + kc0]), 0, 0);
            __builtin_amdgcn_global_load_async_to_lds_b128(
                gp4(A + (size_t)(blockM + row0 + 64) * Kd + kn + kc0),
                lp4(&As[nxt][(row0 + 64) * LDT + kc0]), 0, 0);
            __builtin_amdgcn_global_load_async_to_lds_b128(
                gp4(Bw + (size_t)(blockN + row0) * Kd + kn + kc0),
                lp4(&Bs[nxt][row0 * LDT + kc0]), 0, 0);
            __builtin_amdgcn_global_load_async_to_lds_b128(
                gp4(Bw + (size_t)(blockN + row0 + 64) * Kd + kn + kc0),
                lp4(&Bs[nxt][(row0 + 64) * LDT + kc0]), 0, 0);
            if constexpr (DUAL) {
                __builtin_amdgcn_global_load_async_to_lds_b128(
                    gp4(Bw2 + (size_t)(blockN + row0) * Kd + kn + kc0),
                    lp4(&Bs2[nxt][row0 * LDT + kc0]), 0, 0);
                __builtin_amdgcn_global_load_async_to_lds_b128(
                    gp4(Bw2 + (size_t)(blockN + row0 + 64) * Kd + kn + kc0),
                    lp4(&Bs2[nxt][(row0 + 64) * LDT + kc0]), 0, 0);
            }
        }
#else
        U4 ra0, ra1, rb0, rb1, rc0, rc1;
        if (more) {  // global->reg loads overlap the WMMAs
            ra0 = *reinterpret_cast<const U4*>(A  + (size_t)(blockM + row0) * Kd + kn + kc0);
            ra1 = *reinterpret_cast<const U4*>(A  + (size_t)(blockM + row0 + 64) * Kd + kn + kc0);
            rb0 = *reinterpret_cast<const U4*>(Bw + (size_t)(blockN + row0) * Kd + kn + kc0);
            rb1 = *reinterpret_cast<const U4*>(Bw + (size_t)(blockN + row0 + 64) * Kd + kn + kc0);
            if constexpr (DUAL) {
                rc0 = *reinterpret_cast<const U4*>(Bw2 + (size_t)(blockN + row0) * Kd + kn + kc0);
                rc1 = *reinterpret_cast<const U4*>(Bw2 + (size_t)(blockN + row0 + 64) * Kd + kn + kc0);
            }
        }
#endif

        // -------- fragments + WMMAs on buffer `cur` --------
        Frag af[4], bfr[2], cfr[2];
#pragma unroll
        for (int i = 0; i < 4; ++i) {
            // 16-bit A 16x32: low lanes K 0..7 & 16..23, high lanes K 8..15 & 24..31
            const bf16* p = &As[cur][(wm * 64 + i * 16 + ln) * LDT];
            af[i].q[0] = *reinterpret_cast<const U4*>(p + hlf * 8);
            af[i].q[1] = *reinterpret_cast<const U4*>(p + 16 + hlf * 8);
        }
#pragma unroll
        for (int j = 0; j < 2; ++j) {
            // B 32x16: column = lane&15, low lanes K 0..7 & 16..23, high lanes K 8..15 & 24..31
            const bf16* p = &Bs[cur][(wn * 32 + j * 16 + ln) * LDT];
            bfr[j].q[0] = *reinterpret_cast<const U4*>(p + hlf * 8);
            bfr[j].q[1] = *reinterpret_cast<const U4*>(p + 16 + hlf * 8);
            if constexpr (DUAL) {
                const bf16* p2 = &Bs2[cur][(wn * 32 + j * 16 + ln) * LDT];
                cfr[j].q[0] = *reinterpret_cast<const U4*>(p2 + hlf * 8);
                cfr[j].q[1] = *reinterpret_cast<const U4*>(p2 + 16 + hlf * 8);
            }
        }
#pragma unroll
        for (int i = 0; i < 4; ++i)
#pragma unroll
            for (int j = 0; j < 2; ++j) {
                acc[i][j] = __builtin_amdgcn_wmma_f32_16x16x32_bf16(
                    false, af[i].v, false, bfr[j].v, (short)0, acc[i][j], false, false);
                if constexpr (DUAL)
                    acc2[i][j] = __builtin_amdgcn_wmma_f32_16x16x32_bf16(
                        false, af[i].v, false, cfr[j].v, (short)0, acc2[i][j], false, false);
            }

        if (more) {
#if USE_ASYNC_LDS
            __builtin_amdgcn_s_wait_asynccnt(0);
#else
            *reinterpret_cast<U4*>(&As[nxt][row0 * LDT + kc0]) = ra0;
            *reinterpret_cast<U4*>(&As[nxt][(row0 + 64) * LDT + kc0]) = ra1;
            *reinterpret_cast<U4*>(&Bs[nxt][row0 * LDT + kc0]) = rb0;
            *reinterpret_cast<U4*>(&Bs[nxt][(row0 + 64) * LDT + kc0]) = rb1;
            if constexpr (DUAL) {
                *reinterpret_cast<U4*>(&Bs2[nxt][row0 * LDT + kc0]) = rc0;
                *reinterpret_cast<U4*>(&Bs2[nxt][(row0 + 64) * LDT + kc0]) = rc1;
            }
#endif
            __syncthreads();
        }
        cur = nxt;
    }

    // -------- epilogue: C/D layout — VGPR e: low lanes row=e, high lanes row=e+8
#pragma unroll
    for (int i = 0; i < 4; ++i) {
        const int gm0 = blockM + wm * 64 + i * 16 + hlf * 8;
#pragma unroll
        for (int j = 0; j < 2; ++j) {
            const int gn = blockN + wn * 32 + j * 16 + ln;
            float bi = 0.f;                       // hoisted: invariant over e
            if constexpr (HAS_BIAS) bi = bias[gn];
#pragma unroll
            for (int e = 0; e < 8; ++e) {
                const int row = gm0 + e;
                const size_t idx = (size_t)row * N + gn;
                const float v = acc[i][j][e];
                if constexpr (EPI == EPI_STORE_BF16) {
                    Cb[idx] = (bf16)v;
                } else if constexpr (EPI == EPI_BIAS_GATE_ADD) {
                    Co[idx] += gate[row] * (v + bi);
                } else if constexpr (EPI == EPI_MULSIG_BF16) {
                    const float a = (float)Aux[idx];
                    Cb[idx] = (bf16)(a * sigf(v + bi));
                } else {  // EPI_GLU (DUAL): h = content * sigmoid(gate)
                    const float g = acc2[i][j][e];
                    Cb[idx] = (bf16)(v * sigf(g));
                }
            }
        }
    }
}

// ============================================================================
// host launcher
// ============================================================================
extern "C" void kernel_launch(void* const* d_in, const int* in_sizes, int n_in,
                              void* d_out, int out_size, void* d_ws, size_t ws_size,
                              hipStream_t stream)
{
    const float* x           = (const float*)d_in[0];
    const float* norm_w      = (const float*)d_in[1];
    const float* conv_ws_    = (const float*)d_in[2];
    const float* conv_bs_    = (const float*)d_in[3];
    const float* conv_proj_w = (const float*)d_in[4];
    const float* conv_proj_b = (const float*)d_in[5];
    const float* gate_w      = (const float*)d_in[6];
    const float* router_w    = (const float*)d_in[7];
    const float* router_b    = (const float*)d_in[8];
    const float* head_ws_    = (const float*)d_in[9];
    const float* head_bs_    = (const float*)d_in[10];
    const float* mix_gate_w  = (const float*)d_in[11];
    const float* mix_gate_b  = (const float*)d_in[12];
    const float* mixing_w    = (const float*)d_in[13];
    const float* mixing_b    = (const float*)d_in[14];
    const float* ffn_in_w    = (const float*)d_in[15];
    const float* ffn_out_w   = (const float*)d_in[16];
    const float* pg_w        = (const float*)d_in[17];
    const float* pg_b        = (const float*)d_in[18];
    float* out = (float*)d_out;

    char* w = (char*)d_ws;
    size_t off = 0;
    auto alloc = [&](size_t bytes) -> char* {
        char* p = w + off;
        off = (off + bytes + 255) & ~(size_t)255;
        return p;
    };
    float* n1f    = (float*)alloc((size_t)MTOT * H_ * 4);
    bf16*  n2b    = (bf16*) alloc((size_t)MTOT * H_ * 2);
    bf16*  n3b    = (bf16*) alloc((size_t)MTOT * H_ * 2);
    bf16*  convA  = (bf16*) alloc((size_t)MTOT * H_ * 2);
    bf16*  gbuf   = (bf16*) alloc((size_t)MTOT * 2 * H_ * 2);
    bf16*  state  = (bf16*) alloc((size_t)MTOT * H_ * 2);
    bf16*  zbuf   = (bf16*) alloc((size_t)MTOT * H_ * 2);
    bf16*  hglu   = (bf16*) alloc((size_t)MTOT * INNER_ * 2);
    float* gates  = (float*)alloc((size_t)3 * MTOT * 4);
    float* hwv    = (float*)alloc((size_t)MTOT * NH_ * 4);
    bf16* wb_proj    = (bf16*)alloc((size_t)H_ * H_ * 2);
    bf16* wb_gate    = (bf16*)alloc((size_t)2 * H_ * H_ * 2);
    bf16* wb_mixg    = (bf16*)alloc((size_t)H_ * H_ * 2);
    bf16* wb_mix     = (bf16*)alloc((size_t)H_ * H_ * 2);
    bf16* wb_ffn_in  = (bf16*)alloc((size_t)2 * INNER_ * H_ * 2);
    bf16* wb_ffn_out = (bf16*)alloc((size_t)H_ * INNER_ * 2);

    auto cvt = [&](const float* s, bf16* d, size_t n) {
        const int grid = (int)((n + 256 * 8 - 1) / (256 * 8));
        f32_to_bf16_kernel<<<grid, 256, 0, stream>>>(s, d, n);
    };
    cvt(conv_proj_w, wb_proj, (size_t)H_ * H_);
    cvt(gate_w, wb_gate, (size_t)2 * H_ * H_);
    cvt(mix_gate_w, wb_mixg, (size_t)H_ * H_);
    cvt(mixing_w, wb_mix, (size_t)H_ * H_);
    cvt(ffn_in_w, wb_ffn_in, (size_t)2 * INNER_ * H_);
    cvt(ffn_out_w, wb_ffn_out, (size_t)H_ * INNER_);

    norm_gates_kernel<<<MTOT, 256, 0, stream>>>(
        x, norm_w, pg_w, pg_b, router_w, router_b, out, n1f, n2b, n3b, gates, hwv);

    // ---- conv path ----
    conv_stack_kernel<<<B_ * H_, 256, 0, stream>>>(n1f, conv_ws_, conv_bs_, convA);
    gemm_bf16_k<EPI_BIAS_GATE_ADD, false, true><<<dim3(H_ / BN, MTOT / BM), 256, 0, stream>>>(
        convA, wb_proj, nullptr, MTOT, H_, H_, out, gates + 0 * MTOT, conv_proj_b,
        nullptr, nullptr);

    // ---- state path ----
    gemm_bf16_k<EPI_STORE_BF16, false, false><<<dim3(2 * H_ / BN, MTOT / BM), 256, 0, stream>>>(
        n2b, wb_gate, nullptr, MTOT, 2 * H_, H_, nullptr, nullptr, nullptr, gbuf, nullptr);
    head_conv_kernel<<<B_ * H_, 256, 0, stream>>>(gbuf, head_ws_, head_bs_, hwv, state);
    gemm_bf16_k<EPI_MULSIG_BF16, false, true><<<dim3(H_ / BN, MTOT / BM), 256, 0, stream>>>(
        state, wb_mixg, nullptr, MTOT, H_, H_, nullptr, nullptr, mix_gate_b, zbuf, state);
    gemm_bf16_k<EPI_BIAS_GATE_ADD, false, true><<<dim3(H_ / BN, MTOT / BM), 256, 0, stream>>>(
        zbuf, wb_mix, nullptr, MTOT, H_, H_, out, gates + 1 * MTOT, mixing_b,
        nullptr, nullptr);

    // ---- FFN path: fused dual-B GLU GEMM (content & gate halves share A tile)
    gemm_bf16_k<EPI_GLU, true, false><<<dim3(INNER_ / BN, MTOT / BM), 256, 0, stream>>>(
        n3b, wb_ffn_in, wb_ffn_in + (size_t)INNER_ * H_, MTOT, INNER_, H_,
        nullptr, nullptr, nullptr, hglu, nullptr);
    gemm_bf16_k<EPI_BIAS_GATE_ADD, false, false><<<dim3(H_ / BN, MTOT / BM), 256, 0, stream>>>(
        hglu, wb_ffn_out, nullptr, MTOT, H_, INNER_, out, gates + 2 * MTOT, nullptr,
        nullptr, nullptr);

    (void)in_sizes; (void)n_in; (void)out_size; (void)ws_size;
}
